// SelfAttention_51299089383484
// MI455X (gfx1250) — matrix operand order repaired and checked by reading, not compiled, not run on hardware
//
#include <hip/hip_runtime.h>

// ---------------------------------------------------------------------------
// MHA forward for MI455X (gfx1250): bf16 WMMA, LDS-staged tiles with
// async global->LDS copies, double buffering, flash-attention softmax.
// ---------------------------------------------------------------------------

constexpr int kNB   = 2;
constexpr int kSEQ  = 2048;
constexpr int kEMB  = 1024;
constexpr int kHEAD = 16;
constexpr int kHD   = 64;
constexpr int kM    = kNB * kSEQ;         // 4096

typedef __attribute__((ext_vector_type(16))) __bf16         v16bf;
typedef __attribute__((ext_vector_type(8)))  float          v8f;
typedef __attribute__((ext_vector_type(8)))  unsigned short us8;
typedef int v4i_g __attribute__((__vector_size__(16)));     // matches builtin proto

union Frag16 {
  us8            h[2];
  v16bf          v;
  unsigned short s[16];
};

__device__ __forceinline__ unsigned short f2bf(float f) {
  unsigned int u = __float_as_uint(f);
  u += 0x7FFFu + ((u >> 16) & 1u);        // RNE
  return (unsigned short)(u >> 16);
}

// ---- CDNA5 async global->LDS copy (guarded; fallback = ld + ds_store) -----
#if defined(__AMDGCN__) && __has_builtin(__builtin_amdgcn_global_load_async_to_lds_b128)
#define HAS_ASYNC_CP 1
#else
#define HAS_ASYNC_CP 0
#endif
#if defined(__AMDGCN__) && __has_builtin(__builtin_amdgcn_s_wait_asynccnt)
#define HAS_WAIT_ASYNC 1
#else
#define HAS_WAIT_ASYNC 0
#endif

__device__ __forceinline__ void cp16(void* lds, const void* g) {
#if HAS_ASYNC_CP
  __builtin_amdgcn_global_load_async_to_lds_b128(
      (__attribute__((address_space(1))) v4i_g*)g,
      (__attribute__((address_space(3))) v4i_g*)lds, 0, 0);
#else
  *(us8*)lds = *(const us8*)g;
#endif
}

__device__ __forceinline__ void wait_async_lds() {
#if HAS_WAIT_ASYNC
  __builtin_amdgcn_s_wait_asynccnt(0);
#elif HAS_ASYNC_CP
  asm volatile("s_wait_asynccnt 0x0" ::: "memory");
#endif
}

// ---------------------------------------------------------------------------
__global__ void cvt_f32_bf16(const float* __restrict__ in,
                             unsigned short* __restrict__ out, int n) {
  int i = blockIdx.x * blockDim.x + threadIdx.x;
  if (i < n) out[i] = f2bf(in[i]);
}

// ---------------------------------------------------------------------------
// C[m,n] = sum_k A[m,k] * W[n,k];  A: MxK, W: NxK, both bf16 row-major.
// 256 threads = 8 waves; block tile 128x128; wave tile 64(M)x32(N).
// K-slabs of 32 double-buffered in LDS via async copies.
// ---------------------------------------------------------------------------
constexpr int BK  = 32;
constexpr int LDP = 40;                   // padded LDS row (elems): 80B, 16B-aligned

__global__ __launch_bounds__(256)
void gemm_bt(const unsigned short* __restrict__ A,
             const unsigned short* __restrict__ W,
             unsigned short* __restrict__ Cb,
             float* __restrict__ Cf,
             const float* __restrict__ bias,
             int M, int N, int K, int storeF32) {
  __shared__ unsigned short Asml[2][128 * LDP];
  __shared__ unsigned short Bsml[2][128 * LDP];

  const int tid    = threadIdx.x;
  const int lane   = tid & 31;
  const int wave   = tid >> 5;
  const int lane15 = lane & 15;
  const int hi     = lane >> 4;
  const int khA    = hi * 8;              // A-frag K sel: {0..7,16..23} / +8
  const int khB    = hi * 8;              // B-frag K sel: {0..7,16..23} / +8
  const int wm     = (wave >> 2) * 64;    // 0 / 64
  const int wn     = (wave & 3) * 32;     // 0..96
  const int blockM = blockIdx.x * 128;
  const int blockN = blockIdx.y * 128;

  // cooperative staging: 512 16B-chunks per tile, 2 per thread
  const int r0 = tid >> 2;                // 0..63 (and +64)
  const int p0 = (tid & 3) * 8;           // 8-elem group within 32-K slab

  const unsigned short* ga = A + (size_t)(blockM + r0) * K + p0;
  const unsigned short* gb = W + (size_t)(blockN + r0) * K + p0;

  v8f acc[4][2] = {};

  const int nIter = K / BK;
  // prologue: stage slab 0 into buffer 0
  {
    cp16(&Asml[0][r0 * LDP + p0],        ga);
    cp16(&Asml[0][(r0 + 64) * LDP + p0], ga + (size_t)64 * K);
    cp16(&Bsml[0][r0 * LDP + p0],        gb);
    cp16(&Bsml[0][(r0 + 64) * LDP + p0], gb + (size_t)64 * K);
  }

  for (int it = 0; it < nIter; ++it) {
    wait_async_lds();
    __syncthreads();

    if (it + 1 < nIter) {
      const int buf = (it + 1) & 1;
      const int kk  = (it + 1) * BK;
      cp16(&Asml[buf][r0 * LDP + p0],        ga + kk);
      cp16(&Asml[buf][(r0 + 64) * LDP + p0], ga + kk + (size_t)64 * K);
      cp16(&Bsml[buf][r0 * LDP + p0],        gb + kk);
      cp16(&Bsml[buf][(r0 + 64) * LDP + p0], gb + kk + (size_t)64 * K);
    }

    const unsigned short* As = Asml[it & 1];
    const unsigned short* Bs = Bsml[it & 1];

    Frag16 af[4], bf[2];
#pragma unroll
    for (int i = 0; i < 4; ++i) {
      const unsigned short* p = As + (wm + i * 16 + lane15) * LDP;
      af[i].h[0] = *(const us8*)(p + khA);
      af[i].h[1] = *(const us8*)(p + 16 + khA);
    }
#pragma unroll
    for (int j = 0; j < 2; ++j) {
      const unsigned short* p = Bs + (wn + j * 16 + lane15) * LDP + khB;
      bf[j].h[0] = *(const us8*)(p);
      bf[j].h[1] = *(const us8*)(p + 16);
    }
#pragma unroll
    for (int i = 0; i < 4; ++i)
#pragma unroll
      for (int j = 0; j < 2; ++j)
        acc[i][j] = __builtin_amdgcn_wmma_f32_16x16x32_bf16(
            false, af[i].v, false, bf[j].v, (short)0, acc[i][j], false, false);
  }

#pragma unroll
  for (int i = 0; i < 4; ++i) {
    const int rbase = blockM + wm + i * 16 + hi * 8;
#pragma unroll
    for (int j = 0; j < 2; ++j) {
      const int col = blockN + wn + j * 16 + lane15;
      if (storeF32) {
        const float b = bias[col];
#pragma unroll
        for (int r = 0; r < 8; ++r)
          Cf[(size_t)(rbase + r) * N + col] = acc[i][j][r] + b;
      } else {
#pragma unroll
        for (int r = 0; r < 8; ++r)
          Cb[(size_t)(rbase + r) * N + col] = f2bf(acc[i][j][r]);
      }
    }
  }
}

// ---------------------------------------------------------------------------
// Flash attention: block = 4 waves sharing one (batch, head); each wave owns
// a 16-query tile. K staged async into LDS; V staged transposed into LDS so
// the P@V B-fragment is two contiguous ds_load_b128s.
// scale = 1/sqrt(EMB) = 1/32.
// ---------------------------------------------------------------------------
constexpr int KLP = 72;                   // K-tile padded row (144B, 16B-aligned)
constexpr int VLP = 40;                   // V^T-tile padded row (80B)

__global__ __launch_bounds__(128)
void attn(const unsigned short* __restrict__ Q,
          const unsigned short* __restrict__ Kmat,
          const unsigned short* __restrict__ V,
          unsigned short* __restrict__ O) {
  __shared__ unsigned short Klds[32 * KLP];     // [key][dim]
  __shared__ unsigned short VTlds[64 * VLP];    // [dim][key]
  __shared__ unsigned short Plds[4][16][32];    // per-wave P relayout

  const int tid    = threadIdx.x;
  const int lane   = tid & 31;
  const int wave   = tid >> 5;
  const int lane15 = lane & 15;
  const int hi     = lane >> 4;
  const int khA    = hi * 8;
  const int khB    = hi * 8;

  const int batch = blockIdx.x >> 4;
  const int head  = blockIdx.x & 15;
  const int qbase = (blockIdx.y * 4 + wave) * 16;
  const size_t seq0 = (size_t)batch * kSEQ;

  // staging assignments
  const int kKey  = tid >> 3;             // 0..15 (+16): K tile rows
  const int kPart = (tid & 7) * 8;        // 8-elem group within 64-dim row
  const int vKey  = tid & 31;             // V tile row (key)
  const int vDim0 = (tid >> 5) * 16;      // 16-dim group for transpose

  // Q fragments (d-chunks [0,32) and [32,64)) -- loaded once from global
  const unsigned short* qrow = Q + (seq0 + qbase + lane15) * kEMB + head * kHD;
  Frag16 aq0, aq1;
  aq0.h[0] = *(const us8*)(qrow + khA);
  aq0.h[1] = *(const us8*)(qrow + 16 + khA);
  aq1.h[0] = *(const us8*)(qrow + 32 + khA);
  aq1.h[1] = *(const us8*)(qrow + 48 + khA);

  float mrow[8], lsum[8];
#pragma unroll
  for (int i = 0; i < 8; ++i) { mrow[i] = -3.0e38f; lsum[i] = 0.0f; }
  v8f o[4] = {};

  for (int kb = 0; kb < kSEQ; kb += 32) {
    // ---- stage K tile (async) ----
    const unsigned short* kg =
        Kmat + (seq0 + kb + kKey) * kEMB + head * kHD + kPart;
    cp16(&Klds[kKey * KLP + kPart], kg);
    cp16(&Klds[(kKey + 16) * KLP + kPart], kg + (size_t)16 * kEMB);

    // ---- stage V tile transposed ----
    const unsigned short* vrow =
        V + (seq0 + kb + vKey) * kEMB + head * kHD + vDim0;
    us8 v0 = *(const us8*)(vrow);
    us8 v1 = *(const us8*)(vrow + 8);
#pragma unroll
    for (int j = 0; j < 8; ++j) {
      VTlds[(vDim0 + j) * VLP + vKey]     = v0[j];
      VTlds[(vDim0 + 8 + j) * VLP + vKey] = v1[j];
    }

    wait_async_lds();
    __syncthreads();

    // ---- S = Q @ K^T for 32 keys ----
    v8f s[2] = {};
#pragma unroll
    for (int half = 0; half < 2; ++half) {
      const unsigned short* kr = Klds + (half * 16 + lane15) * KLP;
      Frag16 bk0, bk1;
      bk0.h[0] = *(const us8*)(kr + khB);
      bk0.h[1] = *(const us8*)(kr + khB + 16);
      bk1.h[0] = *(const us8*)(kr + 32 + khB);
      bk1.h[1] = *(const us8*)(kr + 32 + khB + 16);
      s[half] = __builtin_amdgcn_wmma_f32_16x16x32_bf16(
          false, aq0.v, false, bk0.v, (short)0, s[half], false, false);
      s[half] = __builtin_amdgcn_wmma_f32_16x16x32_bf16(
          false, aq1.v, false, bk1.v, (short)0, s[half], false, false);
    }

    // ---- online softmax ----
    float p0[8], p1[8];
#pragma unroll
    for (int i = 0; i < 8; ++i) {
      const float a = s[0][i] * 0.03125f;
      const float b = s[1][i] * 0.03125f;
      float cm = fmaxf(a, b);
      cm = fmaxf(cm, __shfl_xor(cm, 1));
      cm = fmaxf(cm, __shfl_xor(cm, 2));
      cm = fmaxf(cm, __shfl_xor(cm, 4));
      cm = fmaxf(cm, __shfl_xor(cm, 8));
      const float nm   = fmaxf(mrow[i], cm);
      const float corr = __expf(mrow[i] - nm);
      const float e0   = __expf(a - nm);
      const float e1   = __expf(b - nm);
      float rs = e0 + e1;
      rs += __shfl_xor(rs, 1);
      rs += __shfl_xor(rs, 2);
      rs += __shfl_xor(rs, 4);
      rs += __shfl_xor(rs, 8);
      lsum[i] = lsum[i] * corr + rs;
      mrow[i] = nm;
#pragma unroll
      for (int t = 0; t < 4; ++t) o[t][i] *= corr;
      p0[i] = e0; p1[i] = e1;
    }

    // ---- P: C-layout -> A-layout via LDS ----
#pragma unroll
    for (int i = 0; i < 8; ++i) {
      Plds[wave][hi * 8 + i][lane15]      = f2bf(p0[i]);
      Plds[wave][hi * 8 + i][16 + lane15] = f2bf(p1[i]);
    }
    __syncthreads();
    Frag16 pa;
    pa.h[0] = *(const us8*)&Plds[wave][lane15][khA];
    pa.h[1] = *(const us8*)&Plds[wave][lane15][16 + khA];

    // ---- O += P @ V from transposed V tile ----
#pragma unroll
    for (int t = 0; t < 4; ++t) {
      const unsigned short* vr = VTlds + (t * 16 + lane15) * VLP + khB;
      Frag16 bv;
      bv.h[0] = *(const us8*)(vr);
      bv.h[1] = *(const us8*)(vr + 16);
      o[t] = __builtin_amdgcn_wmma_f32_16x16x32_bf16(
          false, pa.v, false, bv.v, (short)0, o[t], false, false);
    }
    __syncthreads();   // protect K/V/P tiles before next iteration's staging
  }

  // ---- normalize and store ----
#pragma unroll
  for (int i = 0; i < 8; ++i) {
    const float inv = 1.0f / lsum[i];
    unsigned short* orow = O + (seq0 + qbase + hi * 8 + i) * kEMB + head * kHD;
#pragma unroll
    for (int t = 0; t < 4; ++t)
      orow[t * 16 + lane15] = f2bf(o[t][i] * inv);
  }
}

// ---------------------------------------------------------------------------
extern "C" void kernel_launch(void* const* d_in, const int* in_sizes, int n_in,
                              void* d_out, int out_size, void* d_ws, size_t ws_size,
                              hipStream_t stream) {
  const float* x  = (const float*)d_in[0];
  const float* Wq = (const float*)d_in[1];
  const float* Wk = (const float*)d_in[2];
  const float* Wv = (const float*)d_in[3];
  const float* Wo = (const float*)d_in[4];
  const float* bo = (const float*)d_in[5];

  const size_t XE = (size_t)kM * kEMB;    // 4M elems
  const size_t WE = (size_t)kEMB * kEMB;  // 1M elems

  unsigned short* Xb  = (unsigned short*)d_ws;
  unsigned short* Wqb = Xb  + XE;
  unsigned short* Wkb = Wqb + WE;
  unsigned short* Wvb = Wkb + WE;
  unsigned short* Wob = Wvb + WE;
  unsigned short* Qb  = Wob + WE;
  unsigned short* Kb  = Qb  + XE;
  unsigned short* Vb  = Kb  + XE;
  unsigned short* Ab  = Vb  + XE;          // ~48 MB total

  cvt_f32_bf16<<<(int)((XE + 255) / 256), 256, 0, stream>>>(x,  Xb,  (int)XE);
  cvt_f32_bf16<<<(int)((WE + 255) / 256), 256, 0, stream>>>(Wq, Wqb, (int)WE);
  cvt_f32_bf16<<<(int)((WE + 255) / 256), 256, 0, stream>>>(Wk, Wkb, (int)WE);
  cvt_f32_bf16<<<(int)((WE + 255) / 256), 256, 0, stream>>>(Wv, Wvb, (int)WE);
  cvt_f32_bf16<<<(int)((WE + 255) / 256), 256, 0, stream>>>(Wo, Wob, (int)WE);

  dim3 gg(kM / 128, kEMB / 128);
  gemm_bt<<<gg, 256, 0, stream>>>(Xb, Wqb, Qb, nullptr, nullptr, kM, kEMB, kEMB, 0);
  gemm_bt<<<gg, 256, 0, stream>>>(Xb, Wkb, Kb, nullptr, nullptr, kM, kEMB, kEMB, 0);
  gemm_bt<<<gg, 256, 0, stream>>>(Xb, Wvb, Vb, nullptr, nullptr, kM, kEMB, kEMB, 0);

  dim3 ga(kNB * kHEAD, kSEQ / 64);
  attn<<<ga, 128, 0, stream>>>(Qb, Kb, Vb, Ab);

  gemm_bt<<<gg, 256, 0, stream>>>(Ab, Wob, nullptr, (float*)d_out, bo, kM, kEMB, kEMB, 1);
}
